// ResBlock_11819749998903
// MI455X (gfx1250) — 8-bit matrix operand order repaired and checked by reading, compiled, not run on hardware
//
#include <hip/hip_runtime.h>
#include <hip/hip_bf16.h>


typedef __attribute__((ext_vector_type(16))) int   v16i;
typedef __attribute__((ext_vector_type(8)))  int   v8i;
typedef __attribute__((ext_vector_type(8)))  float v8f;

#define N_IMG 4
#define C_CH  256
#define HWDIM 96
#define NEG_SLOPE 0.1f
#define BN_EPS 1e-5f
#define WELEMS (C_CH * C_CH * 9)
#define ACT_ELEMS (N_IMG * C_CH * HWDIM * HWDIM)

__device__ __forceinline__ unsigned char fp8_sign(float v) {
    return v > 0.f ? 0x38 : (v < 0.f ? 0xB8 : 0x00);
}

__device__ __forceinline__ int chan_at_pos(int p) {
    int v = (p >> 2) & 7;
    return 32 * (v >> 2) + 16 * ((v >> 1) & 1) + 8 * (p >> 5) + 4 * (v & 1) + (p & 3);
}
__device__ __forceinline__ int chan_store_pos(int k) {
    int v = ((k >> 2) & 1) | (((k >> 4) & 1) << 1) | (((k >> 5) & 1) << 2);
    return ((k >> 3) & 1) * 32 + v * 4 + (k & 3);
}

__global__ void zero_alpha_kernel(float* a) {
    if (threadIdx.x < 2) a[threadIdx.x] = 0.f;
}

__global__ void wabs_reduce_kernel(const float* __restrict__ w, float* __restrict__ sum) {
    __shared__ float red[256];
    float a = 0.f;
    for (int i = blockIdx.x * blockDim.x + threadIdx.x; i < WELEMS;
         i += gridDim.x * blockDim.x)
        a += fabsf(w[i]);
    red[threadIdx.x] = a;
    __syncthreads();
    for (int s = 128; s > 0; s >>= 1) {
        if ((int)threadIdx.x < s) red[threadIdx.x] += red[threadIdx.x + s];
        __syncthreads();
    }
    if (threadIdx.x == 0) atomicAdd(sum, red[0]);
}

__global__ void pack_act_kernel(const float* __restrict__ x, unsigned char* __restrict__ s) {
    int e = blockIdx.x * blockDim.x + threadIdx.x;
    if (e >= ACT_ELEMS) return;
    int cpos = e & 255;
    int pix  = e >> 8;
    int cc   = cpos >> 6;
    int c    = cc * 64 + chan_at_pos(cpos & 63);
    int w = pix % HWDIM;
    int h = (pix / HWDIM) % HWDIM;
    int n = pix / (HWDIM * HWDIM);
    s[e] = fp8_sign(x[(((size_t)n * C_CH + c) * HWDIM + h) * HWDIM + w]);
}

__global__ void pack_wt_kernel(const float* __restrict__ w, unsigned char* __restrict__ p) {
    int e = blockIdx.x * blockDim.x + threadIdx.x;
    if (e >= WELEMS) return;
    int b  = e & 3;
    int j  = (e >> 2) & 15;
    int l  = (e >> 6) & 31;
    int q  = (e >> 11) & 1;
    int tt = e >> 12;
    int t  = tt % 9;
    int ct = tt / 9;
    int k  = (j >> 2) * 32 + ((l & 16) ? 8 : 0) + ((j >> 1) & 1) * 16 + (j & 1) * 4 + b;
    int ci = q * 128 + k;
    int co = ct * 16 + (l & 15);
    int kh = t / 3, kw = t - 3 * kh;
    p[e] = fp8_sign(w[(((size_t)co * C_CH + ci) * 3 + kh) * 3 + kw]);
}

template <int PHASE>
__global__ __launch_bounds__(256, 2)
void binconv_wmma_kernel(const unsigned char* __restrict__ xs,
                         const int* __restrict__ wp,
                         const float* __restrict__ alpha_sum,
                         const float* __restrict__ bias,
                         const float* __restrict__ gamma,
                         const float* __restrict__ beta,
                         const float* __restrict__ mean,
                         const float* __restrict__ var,
                         const float* __restrict__ resid,
                         float* __restrict__ outf,
                         unsigned char* __restrict__ souts) {
    __shared__ __align__(16) unsigned char smem[3 * 34 * 256];

    int bid = blockIdx.x;
    int wt  = bid % 3;
    int h   = (bid / 3) % HWDIM;
    int n   = bid / (3 * HWDIM);
    int w0  = wt * 32;

    int tid = threadIdx.x;
    for (int idx = tid; idx < 3 * 34 * 16; idx += 256) {
        int c16 = idx & 15;
        int pp  = (idx >> 4) % 34;
        int rr  = (idx >> 4) / 34;
        int gh  = h - 1 + rr;
        int gw  = w0 - 1 + pp;
        int4 v = make_int4(0, 0, 0, 0);
        if ((unsigned)gh < HWDIM && (unsigned)gw < HWDIM)
            v = *(const int4*)(xs + (((size_t)(n * HWDIM + gh) * HWDIM + gw) * 256 + c16 * 16));
        *(int4*)(smem + (size_t)idx * 16) = v;
    }
    __syncthreads();

    int wave = tid >> 5;
    int lane = tid & 31;
    int lo   = lane & 15;
    int lh32 = (lane & 16) ? 32 : 0;

    v8f acc00 = {}, acc01 = {}, acc10 = {}, acc11 = {};
    #pragma unroll
    for (int t = 0; t < 9; ++t) {
        const int dh = t / 3, dw = t - 3 * dh;
        #pragma unroll
        for (int q = 0; q < 2; ++q) {
            int base0 = (dh * 34 + (lo + dw)) * 256 + q * 128 + lh32;
            int base1 = base0 + 16 * 256;
            v8i a0l = *(const v8i*)(smem + base0);
            v8i a0h = *(const v8i*)(smem + base0 + 64);
            v8i a1l = *(const v8i*)(smem + base1);
            v8i a1h = *(const v8i*)(smem + base1 + 64);
            v16i A0, A1;
            #pragma unroll
            for (int i = 0; i < 8; ++i) {
                A0[i] = a0l[i]; A0[i + 8] = a0h[i];
                A1[i] = a1l[i]; A1[i + 8] = a1h[i];
            }
            const v16i* bp0 = (const v16i*)
                (wp + ((((size_t)(2 * wave + 0) * 9 + t) * 2 + q) * 32 + lane) * 16);
            const v16i* bp1 = (const v16i*)
                (wp + ((((size_t)(2 * wave + 1) * 9 + t) * 2 + q) * 32 + lane) * 16);
            v16i B0 = *bp0;
            v16i B1 = *bp1;
            acc00 = __builtin_amdgcn_wmma_f32_16x16x128_fp8_fp8(A0, B0, (short)0, acc00, false, false);
            acc01 = __builtin_amdgcn_wmma_f32_16x16x128_fp8_fp8(A0, B1, (short)0, acc01, false, false);
            acc10 = __builtin_amdgcn_wmma_f32_16x16x128_fp8_fp8(A1, B0, (short)0, acc10, false, false);
            acc11 = __builtin_amdgcn_wmma_f32_16x16x128_fp8_fp8(A1, B1, (short)0, acc11, false, false);
        }
    }

    float alpha = alpha_sum[0] * (1.0f / (float)WELEMS);
    int   hi8   = (lane & 16) ? 8 : 0;

    #pragma unroll
    for (int ci = 0; ci < 2; ++ci) {
        int   c    = (2 * wave + ci) * 16 + lo;
        float inv  = gamma[c] * rsqrtf(var[c] + BN_EPS);
        float bnb  = beta[c] - mean[c] * inv;
        float bs   = bias[c];
        int   spos = (c >> 6) * 64 + chan_store_pos(c & 63);
        #pragma unroll
        for (int mi = 0; mi < 2; ++mi) {
            const v8f& acc = (ci == 0) ? (mi == 0 ? acc00 : acc10)
                                       : (mi == 0 ? acc01 : acc11);
            int wbase = w0 + mi * 16 + hi8;
            #pragma unroll
            for (int r = 0; r < 8; ++r) {
                int   wg = wbase + r;
                float y  = (alpha * acc[r] + bs) * inv + bnb;
                size_t pix = ((size_t)n * HWDIM + h) * HWDIM + wg;
                if (PHASE == 1) {
                    y += resid[(((size_t)n * C_CH + c) * HWDIM + h) * HWDIM + wg];
                    y = (y >= 0.f) ? y : NEG_SLOPE * y;
                    outf[pix * 256 + c] = y;
                    souts[pix * 256 + spos] = fp8_sign(y);
                } else {
                    y += resid[pix * 256 + c];
                    outf[(((size_t)n * C_CH + c) * HWDIM + h) * HWDIM + wg] = y;
                }
            }
        }
    }
}

extern "C" void kernel_launch(void* const* d_in, const int* in_sizes, int n_in,
                              void* d_out, int out_size, void* d_ws, size_t ws_size,
                              hipStream_t stream) {
    const float* x   = (const float*)d_in[0];
    const float* w1  = (const float*)d_in[1];
    const float* b1  = (const float*)d_in[2];
    const float* g1  = (const float*)d_in[3];
    const float* be1 = (const float*)d_in[4];
    const float* m1  = (const float*)d_in[5];
    const float* v1  = (const float*)d_in[6];
    const float* w2  = (const float*)d_in[7];
    const float* b2  = (const float*)d_in[8];
    const float* g2  = (const float*)d_in[9];
    const float* be2 = (const float*)d_in[10];
    const float* m2  = (const float*)d_in[11];
    const float* v2  = (const float*)d_in[12];

    char*  ws     = (char*)d_ws;
    float* alphas = (float*)ws;
    unsigned char* xs1 = (unsigned char*)(ws + 256);
    unsigned char* xs2 = xs1 + (size_t)ACT_ELEMS;
    unsigned char* wp1 = xs2 + (size_t)ACT_ELEMS;
    unsigned char* wp2 = wp1 + (size_t)WELEMS;
    float* act1 = (float*)(wp2 + (size_t)WELEMS);

    zero_alpha_kernel<<<1, 32, 0, stream>>>(alphas);
    wabs_reduce_kernel<<<288, 256, 0, stream>>>(w1, alphas + 0);
    wabs_reduce_kernel<<<288, 256, 0, stream>>>(w2, alphas + 1);
    pack_wt_kernel<<<(WELEMS + 255) / 256, 256, 0, stream>>>(w1, wp1);
    pack_wt_kernel<<<(WELEMS + 255) / 256, 256, 0, stream>>>(w2, wp2);
    pack_act_kernel<<<(ACT_ELEMS + 255) / 256, 256, 0, stream>>>(x, xs1);

    const int nblocks = N_IMG * HWDIM * 3;
    binconv_wmma_kernel<1><<<nblocks, 256, 0, stream>>>(
        xs1, (const int*)wp1, alphas + 0, b1, g1, be1, m1, v1, x, act1, xs2);
    binconv_wmma_kernel<2><<<nblocks, 256, 0, stream>>>(
        xs2, (const int*)wp2, alphas + 1, b2, g2, be2, m2, v2, act1,
        (float*)d_out, nullptr);
}
